// SlotAttnHeatmapHead_21560735826007
// MI455X (gfx1250) — matrix operand order repaired and checked by reading, not compiled, not run on hardware
//
#include <hip/hip_runtime.h>

// ---------------------------------------------------------------------------
// SlotAttnHeatmapHead for MI455X (gfx1250), wave32 + WMMA bf16 + TDM.
// Memory-bound (~250MB traffic vs ~30 GFLOP): GEMM weight tiles are DMA'd
// into LDS by the Tensor Data Mover (with pad-insertion for bank-conflict-free
// transposed reads), math runs on v_wmma_f32_16x16x32_bf16.
// ---------------------------------------------------------------------------

typedef __attribute__((ext_vector_type(16))) __bf16 v16bf;
typedef __attribute__((ext_vector_type(8)))  float  v8f;
typedef __attribute__((ext_vector_type(4)))  unsigned int v4u;
typedef __attribute__((ext_vector_type(8)))  int v8i;
typedef __attribute__((ext_vector_type(4)))  int v4i;

#if defined(__has_builtin)
#if __has_builtin(__builtin_amdgcn_tensor_load_to_lds) && \
    __has_builtin(__builtin_amdgcn_s_wait_tensorcnt)
#define HAVE_TDM 1
#endif
#endif

// ---- TDM: 2D f32 tile (tile_w contiguous elems x tile_h rows, row stride
// `stride` elems) global -> LDS, with 4-DWORD pad after every 32 DWORDs so a
// 32-float row lands on a 36-float LDS pitch. One issue per wave (EXEC is
// ignored by TDM) -- caller gates on wave id and waits on TENSORcnt.
__device__ __forceinline__ void tdm_load_2d_f32(const float* g, unsigned lds_off,
                                                int tile_w, int tile_h,
                                                int stride, int td0, int td1) {
#if defined(HAVE_TDM)
  const unsigned long long ga = (unsigned long long)(uintptr_t)g;
  v4u g0;
  g0[0] = 1u;                                   // count=1, user descriptor
  g0[1] = lds_off;                              // lds_addr (bytes)
  g0[2] = (unsigned)ga;                         // global_addr[31:0]
  g0[3] = (unsigned)((ga >> 32) & 0x01FFFFFFu)  // global_addr[56:32]
          | (2u << 30);                         // type = 2 ("image")
  v8i g1;
  g1[0] = (2 << 16)      // data_size = 4B
        | (1 << 20)      // pad_enable
        | (4 << 22)      // pad_interval: every 32 DWORDs
        | (3 << 25);     // pad_amount: 4 DWORDs
  g1[1] = (int)(((unsigned)td0 & 0xffffu) << 16);                 // dim0[15:0]
  g1[2] = (int)((((unsigned)td0 >> 16) & 0xffffu) |
                (((unsigned)td1 & 0xffffu) << 16));               // dim0 hi | dim1 lo
  g1[3] = (int)((((unsigned)td1 >> 16) & 0xffffu) |
                ((unsigned)tile_w << 16));                        // dim1 hi | tile_dim0
  g1[4] = tile_h & 0xffff;                                        // tile_dim1 (dim2=0)
  g1[5] = stride;                                                 // dim0_stride lo
  g1[6] = 0;                                                      // stride hi
  g1[7] = 0;
  v4i z4 = {};
#if __clang_major__ >= 23
  v8i z8 = {};
  __builtin_amdgcn_tensor_load_to_lds(g0, g1, z4, z4, z8, 0);
#else
  __builtin_amdgcn_tensor_load_to_lds(g0, g1, z4, z4, 0);
#endif
#else
  (void)g; (void)lds_off; (void)tile_w; (void)tile_h;
  (void)stride; (void)td0; (void)td1;
#endif
}

// ---- WMMA fragment helpers (layouts per CDNA5 ISA 7.12.2) ------------------

// A fragment 16x32 bf16 from an LDS tile lds[r][k] with `ldl`-float pitch.
// lane r=l&15 -> row m0+r; element i -> K = (i>>3)*16 + (l>>4)*8 + (i&7).
__device__ __forceinline__ v16bf load_a_frag_lds(const float* __restrict__ lds,
                                                 int ldl, int m0, int lane) {
  const int r = lane & 15, hi = lane >> 4;
  const float4* q =
      reinterpret_cast<const float4*>(lds + (size_t)(m0 + r) * ldl + hi * 8);
  float4 l0 = q[0], l1 = q[1];  // K = hi*8 .. +7
  float4 h0 = q[4], h1 = q[5];  // K = 16+hi*8 .. +7
  v16bf a;
  a[0]=(__bf16)l0.x; a[1]=(__bf16)l0.y; a[2]=(__bf16)l0.z; a[3]=(__bf16)l0.w;
  a[4]=(__bf16)l1.x; a[5]=(__bf16)l1.y; a[6]=(__bf16)l1.z; a[7]=(__bf16)l1.w;
  a[8]=(__bf16)h0.x; a[9]=(__bf16)h0.y; a[10]=(__bf16)h0.z; a[11]=(__bf16)h0.w;
  a[12]=(__bf16)h1.x; a[13]=(__bf16)h1.y; a[14]=(__bf16)h1.z; a[15]=(__bf16)h1.w;
  return a;
}

// Same fragment straight from global A (f32 row-major, lda floats).
__device__ __forceinline__ v16bf load_a_frag(const float* __restrict__ A,
                                             int lda, int m0, int k0, int lane) {
  const int r = lane & 15, hi = lane >> 4;
  const float4* q =
      reinterpret_cast<const float4*>(A + (size_t)(m0 + r) * lda + k0 + hi * 8);
  float4 l0 = q[0], l1 = q[1];
  float4 h0 = q[4], h1 = q[5];
  v16bf a;
  a[0]=(__bf16)l0.x; a[1]=(__bf16)l0.y; a[2]=(__bf16)l0.z; a[3]=(__bf16)l0.w;
  a[4]=(__bf16)l1.x; a[5]=(__bf16)l1.y; a[6]=(__bf16)l1.z; a[7]=(__bf16)l1.w;
  a[8]=(__bf16)h0.x; a[9]=(__bf16)h0.y; a[10]=(__bf16)h0.z; a[11]=(__bf16)h0.w;
  a[12]=(__bf16)h1.x; a[13]=(__bf16)h1.y; a[14]=(__bf16)h1.z; a[15]=(__bf16)h1.w;
  return a;
}

// B fragment (NN): 32x16 bf16 from row-major B (K x N, ldb floats).
// lane l -> N column n0+(l&15); element i -> K = k0 + (i>>3)*16 + (l>>4)*8 + (i&7).
__device__ __forceinline__ v16bf load_b_frag_nn(const float* __restrict__ B,
                                                int ldb, int k0, int n0, int lane) {
  const int c = lane & 15, hi = lane >> 4;
  v16bf b;
#pragma unroll
  for (int i = 0; i < 16; ++i)
    b[i] = (__bf16)B[(size_t)(k0 + (i >> 3) * 16 + hi * 8 + (i & 7)) * ldb + n0 + c];
  return b;
}

// B fragment from an LDS W-tile lds[n][k] (ldl-float pitch):
// element i -> lds[nloc+(lane&15)][(i>>3)*16 + (lane>>4)*8 + (i&7)].
__device__ __forceinline__ v16bf load_b_frag_lds_t(const float* __restrict__ lds,
                                                   int ldl, int nloc, int lane) {
  v16bf b;
#pragma unroll
  for (int i = 0; i < 16; ++i)
    b[i] = (__bf16)lds[(size_t)(nloc + (lane & 15)) * ldl + (i >> 3) * 16 + (lane >> 4) * 8 + (i & 7)];
  return b;
}

// C/D: vgpr v -> row m0 + (lane>>4)*8 + v, col n0 + (lane&15).
template <int ACT>  // 0 = none, 1 = SiLU
__device__ __forceinline__ void store_tile(float* __restrict__ C, int ldc,
                                           int m0, int n0, v8f acc,
                                           const float* __restrict__ bias,
                                           float alpha, int lane) {
  const int r = lane & 15, hi = lane >> 4;
  const int col = n0 + r;
  const float bv = bias ? bias[col] : 0.f;
#pragma unroll
  for (int v = 0; v < 8; ++v) {
    float x = acc[v] * alpha + bv;
    if (ACT == 1) x = x / (1.f + __expf(-x));
    C[(size_t)(m0 + hi * 8 + v) * ldc + col] = x;
  }
}

// ---- GEMM NT: C[m,n] = act(alpha * sum_k A[m,k] * W[n,k] + bias[n]) --------
// Tile BM=32, BN=128 (8 waves x 16 cols), BK=32. Both A and W tiles are
// DMA'd into LDS by the TDM (pitch 36 via pad descriptor); fallback is async
// global->LDS copies. Requires M%32==0, N%128==0, K%32==0.
template <int ACT>
__global__ void __launch_bounds__(256)
gemm_nt_wmma(const float* __restrict__ A, int lda,
             const float* __restrict__ W, int ldw,
             const float* __restrict__ bias,
             float* __restrict__ C, int ldc,
             int M, int N, int Kd, float alpha) {
  constexpr int BM = 32, BN = 128, BK = 32, LDL = BK + 4;  // LDS pitch 36
  __shared__ float lds_w[BN * LDL];  // lds_w[n][k]
  __shared__ float lds_a[BM * LDL];  // lds_a[m][k]

  const int tid = threadIdx.x, lane = tid & 31, wave = tid >> 5;
  const int n0 = blockIdx.x * BN;
  const int m0 = blockIdx.y * BM;

  v8f acc0 = {}, acc1 = {};

  for (int kk = 0; kk < Kd; kk += BK) {
#if defined(HAVE_TDM)
    if (wave == 0) {  // TDM ignores EXEC; issue once per workgroup
      tdm_load_2d_f32(W + (size_t)n0 * ldw + kk,
                      (unsigned)(uintptr_t)&lds_w[0], BK, BN, ldw, ldw, N - n0);
      tdm_load_2d_f32(A + (size_t)m0 * lda + kk,
                      (unsigned)(uintptr_t)&lds_a[0], BK, BM, lda, lda, M - m0);
      __builtin_amdgcn_s_wait_tensorcnt(0);
    }
#else
    // async global->LDS copies (ASYNCcnt), b128 chunks, manual pad pitch
    for (int c = tid; c < BN * (BK / 4); c += 256) {
      const int n = c >> 3, j = c & 7;
      const float* src = W + (size_t)(n0 + n) * ldw + kk + j * 4;
      const unsigned loff =
          (unsigned)(uintptr_t)&lds_w[0] + (unsigned)((n * LDL + j * 4) * 4);
      asm volatile("global_load_async_to_lds_b128 %0, %1, off"
                   :: "v"(loff), "v"((unsigned long long)(uintptr_t)src)
                   : "memory");
    }
    for (int c = tid; c < BM * (BK / 4); c += 256) {
      const int m = c >> 3, j = c & 7;
      const float* src = A + (size_t)(m0 + m) * lda + kk + j * 4;
      const unsigned loff =
          (unsigned)(uintptr_t)&lds_a[0] + (unsigned)((m * LDL + j * 4) * 4);
      asm volatile("global_load_async_to_lds_b128 %0, %1, off"
                   :: "v"(loff), "v"((unsigned long long)(uintptr_t)src)
                   : "memory");
    }
    asm volatile("s_wait_asynccnt 0x0" ::: "memory");
#endif
    __syncthreads();

    v16bf a0 = load_a_frag_lds(lds_a, LDL, 0, lane);
    v16bf a1 = load_a_frag_lds(lds_a, LDL, 16, lane);
    v16bf bf = load_b_frag_lds_t(lds_w, LDL, wave * 16, lane);
    acc0 = __builtin_amdgcn_wmma_f32_16x16x32_bf16(false, a0, false, bf,
                                                   (short)0, acc0, false, false);
    acc1 = __builtin_amdgcn_wmma_f32_16x16x32_bf16(false, a1, false, bf,
                                                   (short)0, acc1, false, false);
    __syncthreads();
  }
  store_tile<ACT>(C, ldc, m0, n0 + wave * 16, acc0, bias, alpha, lane);
  store_tile<ACT>(C, ldc, m0 + 16, n0 + wave * 16, acc1, bias, alpha, lane);
}

// ---- GEMM NN (batched): C[z][m,n] = alpha * sum_k A[z][m,k] * B[z][k,n] ----
// B rows are N-contiguous -> direct 4-byte loads, coalesced across 16 lanes.
__global__ void __launch_bounds__(256)
gemm_nn_wmma(const float* __restrict__ A, int lda, long sA,
             const float* __restrict__ B, int ldb, long sB,
             float* __restrict__ C, int ldc, long sC,
             int M, int N, int Kd, float alpha) {
  const int z = blockIdx.z;
  const float* Ab = A + (size_t)z * sA;
  const float* Bb = B + (size_t)z * sB;
  float* Cb = C + (size_t)z * sC;

  const int lane = threadIdx.x & 31, wave = threadIdx.x >> 5;
  const int nw = blockIdx.x * 128 + wave * 16;
  const int m0 = blockIdx.y * 32;

  v8f acc0 = {}, acc1 = {};
  for (int kk = 0; kk < Kd; kk += 32) {
    if (kk + 32 < Kd)  // pull next B rows toward GL2 (global_prefetch_b8)
      __builtin_prefetch(Bb + (size_t)(kk + 32 + lane) * ldb + nw, 0, 1);
    v16bf a0 = load_a_frag(Ab, lda, m0, kk, lane);
    v16bf a1 = load_a_frag(Ab, lda, m0 + 16, kk, lane);
    v16bf bf = load_b_frag_nn(Bb, ldb, kk, nw, lane);
    acc0 = __builtin_amdgcn_wmma_f32_16x16x32_bf16(false, a0, false, bf,
                                                   (short)0, acc0, false, false);
    acc1 = __builtin_amdgcn_wmma_f32_16x16x32_bf16(false, a1, false, bf,
                                                   (short)0, acc1, false, false);
  }
  store_tile<0>(Cb, ldc, m0, nw, acc0, nullptr, alpha, lane);
  store_tile<0>(Cb, ldc, m0 + 16, nw, acc1, nullptr, alpha, lane);
}

// ---- Attention logits: L[b,k,h,xy] = (1/8) * sum_d Q[b,k,h,d]*Kmap[b,h,d,xy]
__global__ void __launch_bounds__(256)
attn_logits_wmma(const float* __restrict__ Q,     // (8,32,4,64)
                 const float* __restrict__ Kmap,  // (8,4,64,4096)
                 float* __restrict__ Lg) {        // (8,32,4,4096)
  const int z = blockIdx.z, b = z >> 2, h = z & 3;
  const float* Ab = Q + (size_t)b * 32 * 256 + h * 64;           // lda=256
  const float* Bb = Kmap + (size_t)z * 64 * 4096;                // ldb=4096
  float* Cb = Lg + (size_t)b * 32 * 4 * 4096 + (size_t)h * 4096; // ldc=16384

  const int lane = threadIdx.x & 31, wave = threadIdx.x >> 5;
  const int nw = blockIdx.x * 128 + wave * 16;

  v8f acc0 = {}, acc1 = {};
#pragma unroll
  for (int kk = 0; kk < 64; kk += 32) {
    v16bf a0 = load_a_frag(Ab, 256, 0, kk, lane);
    v16bf a1 = load_a_frag(Ab, 256, 16, kk, lane);
    v16bf bf = load_b_frag_nn(Bb, 4096, kk, nw, lane);
    acc0 = __builtin_amdgcn_wmma_f32_16x16x32_bf16(false, a0, false, bf,
                                                   (short)0, acc0, false, false);
    acc1 = __builtin_amdgcn_wmma_f32_16x16x32_bf16(false, a1, false, bf,
                                                   (short)0, acc1, false, false);
  }
  store_tile<0>(Cb, 4 * 4096, 0, nw, acc0, nullptr, 0.125f, lane);
  store_tile<0>(Cb, 4 * 4096, 16, nw, acc1, nullptr, 0.125f, lane);
}

// ---- Row LayerNorm over [src0 | src1] (src1 optional) ----------------------
__global__ void __launch_bounds__(256)
ln_rows(const float* __restrict__ src0, const float* __restrict__ src1,
        int w0, int w1, const float* __restrict__ g,
        const float* __restrict__ be, float* __restrict__ out) {
  const int row = blockIdx.x, tid = threadIdx.x;
  const int Wd = w0 + w1;
  const float* p0 = src0 + (size_t)row * w0;
  const float* p1 = src1 ? src1 + (size_t)row * w1 : nullptr;

  float s = 0.f, s2 = 0.f;
  for (int i = tid; i < w0; i += 256) { float v = p0[i]; s += v; s2 += v * v; }
  if (p1)
    for (int i = tid; i < w1; i += 256) { float v = p1[i]; s += v; s2 += v * v; }

  for (int o = 16; o > 0; o >>= 1) {
    s += __shfl_down(s, o, 32);
    s2 += __shfl_down(s2, o, 32);
  }
  __shared__ float rs_[8], rs2_[8], mv[2];
  const int lane = tid & 31, wave = tid >> 5;
  if (lane == 0) { rs_[wave] = s; rs2_[wave] = s2; }
  __syncthreads();
  if (tid == 0) {
    float t = 0.f, t2 = 0.f;
    for (int i = 0; i < 8; ++i) { t += rs_[i]; t2 += rs2_[i]; }
    const float mu = t / (float)Wd;
    mv[0] = mu;
    mv[1] = rsqrtf(t2 / (float)Wd - mu * mu + 1e-5f);
  }
  __syncthreads();
  const float mu = mv[0], rstd = mv[1];
  float* op = out + (size_t)row * Wd;
  for (int i = tid; i < w0; i += 256)
    op[i] = (p0[i] - mu) * rstd * g[i] + be[i];
  if (p1)
    for (int i = tid; i < w1; i += 256)
      op[w0 + i] = (p1[i] - mu) * rstd * g[w0 + i] + be[w0 + i];
}

// ---- Fused 3x3 conv head: (4->4 SiLU) then (4->1), one map per block ------
__global__ void __launch_bounds__(256)
fuser_conv(const float* __restrict__ X,   // (256,4,64,64)
           const float* __restrict__ w1,  // (4,4,3,3)
           const float* __restrict__ b1,  // (4)
           const float* __restrict__ w2,  // (1,4,3,3)
           const float* __restrict__ b2,  // (1)
           float* __restrict__ Y) {       // (256,64,64)
  __shared__ float wt1[4][4][3][3];
  __shared__ float wt2[4][3][3];
  __shared__ float xin[4][20][64];
  __shared__ float h1[4][18][64];
  const int map = blockIdx.x, tid = threadIdx.x;
  if (tid < 144) (&wt1[0][0][0][0])[tid] = w1[tid];
  if (tid < 36) (&wt2[0][0][0])[tid] = w2[tid];
  const float* xp = X + (size_t)map * 4 * 4096;
  float* yp = Y + (size_t)map * 4096;

  for (int y0 = 0; y0 < 64; y0 += 16) {
    __syncthreads();
    for (int i = tid; i < 4 * 20 * 64; i += 256) {
      const int c = i / (20 * 64), rr = (i / 64) % 20, x = i & 63;
      const int y = y0 - 2 + rr;
      xin[c][rr][x] = (y >= 0 && y < 64) ? xp[c * 4096 + y * 64 + x] : 0.f;
    }
    __syncthreads();
    for (int i = tid; i < 4 * 18 * 64; i += 256) {
      const int oc = i / (18 * 64), rr = (i / 64) % 18, x = i & 63;
      const int y = y0 - 1 + rr;
      float v = 0.f;
      if (y >= 0 && y < 64) {
        float acc = b1[oc];
#pragma unroll
        for (int ic = 0; ic < 4; ++ic)
#pragma unroll
          for (int dy = 0; dy < 3; ++dy) {
            const int xr = rr + dy;
#pragma unroll
            for (int dx = 0; dx < 3; ++dx) {
              const int xx = x + dx - 1;
              if (xx >= 0 && xx < 64) acc += wt1[oc][ic][dy][dx] * xin[ic][xr][xx];
            }
          }
        v = acc / (1.f + __expf(-acc));
      }
      h1[oc][rr][x] = v;
    }
    __syncthreads();
    for (int i = tid; i < 16 * 64; i += 256) {
      const int rr = i >> 6, x = i & 63;
      float acc = b2[0];
#pragma unroll
      for (int ic = 0; ic < 4; ++ic)
#pragma unroll
        for (int dy = 0; dy < 3; ++dy) {
          const int hr = rr + dy;
#pragma unroll
          for (int dx = 0; dx < 3; ++dx) {
            const int xx = x + dx - 1;
            if (xx >= 0 && xx < 64) acc += wt2[ic][dy][dx] * h1[ic][hr][xx];
          }
        }
      yp[(y0 + rr) * 64 + x] = acc;
    }
  }
}

// ---- Bilinear 4x upsample 64->256, half-pixel centers; NT streaming store --
__global__ void __launch_bounds__(256)
upsample4(const float* __restrict__ L, float* __restrict__ O) {
  const int map = blockIdx.y, oy = blockIdx.x, ox = threadIdx.x;
  const float sy = 0.25f * oy - 0.375f, sx = 0.25f * ox - 0.375f;
  const int yf = (int)floorf(sy), xf = (int)floorf(sx);
  const float wy = sy - yf, wx = sx - xf;
  const int y0 = min(max(yf, 0), 63), y1 = min(max(yf + 1, 0), 63);
  const int x0 = min(max(xf, 0), 63), x1 = min(max(xf + 1, 0), 63);
  const float* p = L + (size_t)map * 4096;
  const float v00 = p[y0 * 64 + x0], v01 = p[y0 * 64 + x1];
  const float v10 = p[y1 * 64 + x0], v11 = p[y1 * 64 + x1];
  const float v = (1.f - wy) * ((1.f - wx) * v00 + wx * v01) +
                  wy * ((1.f - wx) * v10 + wx * v11);
  __builtin_nontemporal_store(v, O + (size_t)map * 65536 + oy * 256 + ox);
}

// ---- Confidence head final dot: conf[m] = cfh[m,:] . w2 + b2 ---------------
__global__ void __launch_bounds__(256)
conf_head(const float* __restrict__ cfh, const float* __restrict__ w2,
          const float* __restrict__ b2, float* __restrict__ out) {
  const int row = blockIdx.x, tid = threadIdx.x;
  float s = 0.f;
  for (int i = tid; i < 2048; i += 256) s += cfh[(size_t)row * 2048 + i] * w2[i];
  for (int o = 16; o > 0; o >>= 1) s += __shfl_down(s, o, 32);
  __shared__ float red[8];
  const int lane = tid & 31, wave = tid >> 5;
  if (lane == 0) red[wave] = s;
  __syncthreads();
  if (tid == 0) {
    float t = 0.f;
    for (int i = 0; i < 8; ++i) t += red[i];
    out[row] = t + b2[0];
  }
}

// ---------------------------------------------------------------------------
extern "C" void kernel_launch(void* const* d_in, const int* in_sizes, int n_in,
                              void* d_out, int out_size, void* d_ws,
                              size_t ws_size, hipStream_t stream) {
  (void)in_sizes; (void)n_in; (void)out_size; (void)ws_size;
  const float* image_embed = (const float*)d_in[0];   // (8,256,64,64)
  const float* ref_vec     = (const float*)d_in[1];   // (8,32,4096)
  const float* querybook   = (const float*)d_in[2];   // (8,32,512)
  const float* q_proj_w    = (const float*)d_in[3];   // (256,4096)
  const float* k_proj_w    = (const float*)d_in[4];   // (256,256,1,1)
  const float* fuser_w1    = (const float*)d_in[5];   // (4,4,3,3)
  const float* fuser_b1    = (const float*)d_in[6];   // (4)
  const float* fuser_w2    = (const float*)d_in[7];   // (1,4,3,3)
  const float* fuser_b2    = (const float*)d_in[8];   // (1)
  const float* qb_proj_w   = (const float*)d_in[9];   // (4096,512)
  const float* rf_ln_g     = (const float*)d_in[10];  // (8192)
  const float* rf_ln_b     = (const float*)d_in[11];  // (8192)
  const float* rf_lin_w    = (const float*)d_in[12];  // (4096,8192)
  const float* rf_lin_b    = (const float*)d_in[13];  // (4096)
  const float* cf_ln_g     = (const float*)d_in[14];  // (4096)
  const float* cf_ln_b     = (const float*)d_in[15];  // (4096)
  const float* cf_w1       = (const float*)d_in[16];  // (2048,4096)
  const float* cf_b1       = (const float*)d_in[17];  // (2048)
  const float* cf_w2       = (const float*)d_in[18];  // (1,2048)
  const float* cf_b2       = (const float*)d_in[19];  // (1)

  float* out = (float*)d_out;  // 8*32*256*256 heatmaps then 8*32 conf

  // workspace carve-up (~78 MB)
  float* qb    = (float*)d_ws;                 // 256*4096
  float* catn  = qb    + (size_t)256 * 4096;   // 256*8192
  float* fused = catn  + (size_t)256 * 8192;   // 256*4096
  float* fln   = fused + (size_t)256 * 4096;   // 256*4096
  float* Qm    = fln   + (size_t)256 * 4096;   // 256*256
  float* Kmap  = Qm    + (size_t)256 * 256;    // 8*256*4096
  float* lgH   = Kmap  + (size_t)8 * 256 * 4096;  // 256*4*4096
  float* lg64  = lgH   + (size_t)256 * 4 * 4096;  // 256*4096
  float* cfh   = lg64  + (size_t)256 * 4096;      // 256*2048

  const dim3 blk(256);

  // 1) qb = querybook (256,512) x qb_proj_w^T (4096,512)
  gemm_nt_wmma<0><<<dim3(4096 / 128, 256 / 32), blk, 0, stream>>>(
      querybook, 512, qb_proj_w, 512, nullptr, qb, 4096, 256, 4096, 512, 1.f);

  // 2) LN over concat [ref_vec | qb] (width 8192)
  ln_rows<<<256, blk, 0, stream>>>(ref_vec, qb, 4096, 4096, rf_ln_g, rf_ln_b,
                                   catn);

  // 3) fused = SiLU(catn x rf_lin_w^T + b)   (256,4096), K=8192
  gemm_nt_wmma<1><<<dim3(4096 / 128, 256 / 32), blk, 0, stream>>>(
      catn, 8192, rf_lin_w, 8192, rf_lin_b, fused, 4096, 256, 4096, 8192, 1.f);

  // 4) Q = fused x q_proj_w^T   (256,256)
  gemm_nt_wmma<0><<<dim3(256 / 128, 256 / 32), blk, 0, stream>>>(
      fused, 4096, q_proj_w, 4096, nullptr, Qm, 256, 256, 256, 4096, 1.f);

  // 5) Kmap[b] = k_proj_w (256,256) x image_embed[b] (256,4096)  (batched NN)
  gemm_nn_wmma<<<dim3(4096 / 128, 256 / 32, 8), blk, 0, stream>>>(
      k_proj_w, 256, 0, image_embed, 4096, (long)256 * 4096, Kmap, 4096,
      (long)256 * 4096, 256, 4096, 256, 1.f);

  // 6) logits[b,k,h,xy] = Q . Kmap / sqrt(64)
  attn_logits_wmma<<<dim3(4096 / 128, 1, 32), blk, 0, stream>>>(Qm, Kmap, lgH);

  // 7) fused conv head (3x3 SiLU, 3x3) -> logits64 (256,64,64)
  fuser_conv<<<256, blk, 0, stream>>>(lgH, fuser_w1, fuser_b1, fuser_w2,
                                      fuser_b2, lg64);

  // 8) bilinear 4x upsample -> d_out heatmaps (8,32,256,256)
  upsample4<<<dim3(256, 256), blk, 0, stream>>>(lg64, out);

  // 9) LN of fused (width 4096)
  ln_rows<<<256, blk, 0, stream>>>(fused, nullptr, 4096, 0, cf_ln_g, cf_ln_b,
                                   fln);

  // 10) cfh = SiLU(fln x cf_w1^T + b1)   (256,2048), K=4096
  gemm_nt_wmma<1><<<dim3(2048 / 128, 256 / 32), blk, 0, stream>>>(
      fln, 4096, cf_w1, 4096, cf_b1, cfh, 2048, 256, 2048, 4096, 1.f);

  // 11) conf logits -> tail of d_out
  conf_head<<<256, blk, 0, stream>>>(cfh, cf_w2, cf_b2,
                                     out + (size_t)8 * 32 * 256 * 256);
}
